// UncertainNetwork_66340064854506
// MI455X (gfx1250) — compile-verified, matrix operand order checked by reading, not run on hardware
//
#include <hip/hip_runtime.h>
#include <math.h>

#define DEV __device__ __forceinline__

typedef __attribute__((ext_vector_type(16))) __bf16 v16bf;
typedef __attribute__((ext_vector_type(8)))  float  v8f;

constexpr int Bc = 2, Sc = 512, Dc = 768, Hc = 12, DFFc = 3072;
constexpr int VOCABc = 50257, DIc = 1536, NINDc = 5, HDc = 64;
constexpr int BSc = Bc * Sc;   // 1024 tokens
constexpr float EPSc = 1e-5f;

enum { EPI_NONE = 0, EPI_GELU = 1, EPI_SOFTPLUS = 2 };

DEV float sigm(float x)  { return 1.f / (1.f + expf(-x)); }
DEV float siluf(float x) { return x * sigm(x); }
DEV float geluf(float x) {
  float x3 = x * x * x;
  return 0.5f * x * (1.f + tanhf(0.7978845608028654f * (x + 0.044715f * x3)));
}
DEV float softplusf(float x) { return (x > 20.f) ? x : log1pf(expf(x)); }

DEV unsigned bfbits(float f) {            // RTNE fp32 -> bf16 bits
  unsigned u = __builtin_bit_cast(unsigned, f);
  return (u + 0x7FFFu + ((u >> 16) & 1u)) >> 16;
}
DEV __bf16 f2bf(float f) {
  unsigned short s = (unsigned short)bfbits(f);
  return __builtin_bit_cast(__bf16, s);
}
DEV unsigned pack2(float a, float b) { return bfbits(a) | (bfbits(b) << 16); }

// ---------------------------------------------------------------------------
// Generic GEMM: C[M,N] = epi(A[M,K] @ W[K,N] + bias), bf16 WMMA, fp32 accum.
// 128x64 tile / 256-thread block; 8 waves in 4(M) x 2(N) grid; each wave owns
// a 32x32 slab -> 4 v_wmma_f32_16x16x32_bf16 per K-step of 32.
// LDS layouts are K-contiguous so every fragment is two ds_load_b128.
// ---------------------------------------------------------------------------
__global__ __launch_bounds__(256) void gemm_bf16_wmma(
    const float* __restrict__ A, int lda,
    const float* __restrict__ W, int ldw,
    const float* __restrict__ bias,
    float* __restrict__ C, int ldc,
    int M, int N, int K, int epi)
{
  __shared__ __bf16 As[128][32];   // [row][k]   (K contiguous)
  __shared__ __bf16 Bt[64][32];    // [col][k]   (transposed, K contiguous)

  const int tid  = threadIdx.x;
  const int wave = tid >> 5;
  const int lane = tid & 31;
  const int m0   = blockIdx.y * 128;
  const int n0   = blockIdx.x * 64;
  const int wr   = (wave & 3) * 32;     // wave row base within tile
  const int wc   = (wave >> 2) * 32;    // wave col base within tile
  const int l16  = lane & 15;
  const int grp  = lane >> 4;

  const bool full = (m0 + 128 <= M) && (n0 + 64 <= N);

  union Frag { v16bf v; __bf16 e[16]; uint4 q[2]; };
  v8f acc00 = {}, acc01 = {}, acc10 = {}, acc11 = {};

  for (int k0 = 0; k0 < K; k0 += 32) {
    if (full) {
      // A tile: 128x32 floats = 1024 float4, 4 per thread, K-contiguous rows
#pragma unroll
      for (int e = 0; e < 4; ++e) {
        int i4 = tid + e * 256;
        int r  = i4 >> 3;            // 8 float4 per row
        int kk = (i4 & 7) * 4;
        float4 v = *(const float4*)&A[(size_t)(m0 + r) * lda + k0 + kk];
        uint2 p; p.x = pack2(v.x, v.y); p.y = pack2(v.z, v.w);
        *(uint2*)&As[r][kk] = p;
      }
      // B tile: 32x64 floats = 512 float4, 2 per thread; store transposed
#pragma unroll
      for (int e = 0; e < 2; ++e) {
        int i4 = tid + e * 256;
        int kk = i4 >> 4;            // 16 float4 per k-row
        int cc = (i4 & 15) * 4;
        float4 v = *(const float4*)&W[(size_t)(k0 + kk) * ldw + n0 + cc];
        Bt[cc + 0][kk] = f2bf(v.x);
        Bt[cc + 1][kk] = f2bf(v.y);
        Bt[cc + 2][kk] = f2bf(v.z);
        Bt[cc + 3][kk] = f2bf(v.w);
      }
    } else {
      // ragged fallback (only the last block column of the Wx GEMM)
#pragma unroll
      for (int e = 0; e < 16; ++e) {
        int i = tid + e * 256;
        int r = i >> 5, kk = i & 31;
        int gm = m0 + r, gk = k0 + kk;
        float v = (gm < M && gk < K) ? A[(size_t)gm * lda + gk] : 0.f;
        As[r][kk] = f2bf(v);
      }
#pragma unroll
      for (int e = 0; e < 8; ++e) {
        int i = tid + e * 256;
        int kk = i >> 6, cc = i & 63;
        int gk = k0 + kk, gn = n0 + cc;
        float v = (gk < K && gn < N) ? W[(size_t)gk * ldw + gn] : 0.f;
        Bt[cc][kk] = f2bf(v);
      }
    }
    // prefetch next K tiles into GL2 (global_prefetch_b8)
    if (full && k0 + 32 < K) {
      __builtin_prefetch(&A[(size_t)(m0 + (tid >> 1)) * lda + k0 + 32 + (tid & 1) * 16], 0, 1);
      if (tid < 128)
        __builtin_prefetch(&W[(size_t)(k0 + 32 + (tid >> 2)) * ldw + n0 + (tid & 3) * 16], 0, 1);
    }
    __syncthreads();

    // fragments: elements 0..7 <-> K = grp*8..+7; 8..15 <-> K = 16+grp*8..+7
    // -> two contiguous 16B LDS loads each (ds_load_b128)
    Frag a0, a1, b0, b1;
    const int ka = grp * 8;
    a0.q[0] = *(const uint4*)&As[wr + l16][ka];
    a0.q[1] = *(const uint4*)&As[wr + l16][16 + ka];
    a1.q[0] = *(const uint4*)&As[wr + 16 + l16][ka];
    a1.q[1] = *(const uint4*)&As[wr + 16 + l16][16 + ka];
    b0.q[0] = *(const uint4*)&Bt[wc + l16][ka];
    b0.q[1] = *(const uint4*)&Bt[wc + l16][16 + ka];
    b1.q[0] = *(const uint4*)&Bt[wc + 16 + l16][ka];
    b1.q[1] = *(const uint4*)&Bt[wc + 16 + l16][16 + ka];
#if defined(__HIP_DEVICE_COMPILE__)
    acc00 = __builtin_amdgcn_wmma_f32_16x16x32_bf16(false, a0.v, false, b0.v, (short)0, acc00, false, false);
    acc01 = __builtin_amdgcn_wmma_f32_16x16x32_bf16(false, a0.v, false, b1.v, (short)0, acc01, false, false);
    acc10 = __builtin_amdgcn_wmma_f32_16x16x32_bf16(false, a1.v, false, b0.v, (short)0, acc10, false, false);
    acc11 = __builtin_amdgcn_wmma_f32_16x16x32_bf16(false, a1.v, false, b1.v, (short)0, acc11, false, false);
#endif
    __syncthreads();
  }

  // epilogue: D layout -> lane col = l16; VGPR r holds M = base + grp*8 + r
#pragma unroll
  for (int rt = 0; rt < 2; ++rt) {
#pragma unroll
    for (int ct = 0; ct < 2; ++ct) {
      const v8f& a = rt ? (ct ? acc11 : acc10) : (ct ? acc01 : acc00);
      int mb = m0 + wr + rt * 16 + grp * 8;
      int nn = n0 + wc + ct * 16 + l16;
      if (nn >= N) continue;
      float bv = bias ? bias[nn] : 0.f;
#pragma unroll
      for (int r = 0; r < 8; ++r) {
        int m = mb + r;
        if (m >= M) continue;
        float v = a[r] + bv;
        if (epi == EPI_GELU)          v = geluf(v);
        else if (epi == EPI_SOFTPLUS) v = softplusf(v);
        C[(size_t)m * ldc + nn] = v;
      }
    }
  }
}

// ---------------------------------------------------------------------------
// Rotary embedding applied to the input embeddings.
// ---------------------------------------------------------------------------
__global__ void rotary_kernel(const float* __restrict__ x, float* __restrict__ out) {
  int idx = blockIdx.x * blockDim.x + threadIdx.x;
  if (idx >= BSc * Dc) return;
  int i = idx & 63;                 // pos within head (D multiple of 64)
  int s = (idx / Dc) % Sc;
  int j = i & 31;
  float inv = expf(-((float)(2 * j) / (float)HDc) * 9.210340371976184f); // ln 1e4
  float f = (float)s * inv;
  float c = cosf(f), sn = sinf(f);
  float x1 = x[idx];
  float rot = (i < 32) ? -x[idx + 32] : x[idx - 32];
  out[idx] = x1 * c + rot * sn;
}

// ---------------------------------------------------------------------------
// Depthwise causal conv (DCONV=4) over xi = xz[:, :1536], + bias, silu.
// ---------------------------------------------------------------------------
__global__ void conv_silu_kernel(const float* __restrict__ xz,
                                 const float* __restrict__ w,
                                 const float* __restrict__ cb,
                                 float* __restrict__ xc) {
  int idx = blockIdx.x * blockDim.x + threadIdx.x;
  if (idx >= BSc * DIc) return;
  int c = idx % DIc;
  int bs = idx / DIc;
  int s = bs % Sc, b = bs / Sc;
  float acc = cb[c];
#pragma unroll
  for (int t = 0; t < 4; ++t) {
    int sp = s - 3 + t;
    if (sp >= 0) acc += w[c * 4 + t] * xz[((size_t)(b * Sc + sp)) * 3072 + c];
  }
  xc[idx] = siluf(acc);
}

// ---------------------------------------------------------------------------
// Selective scan: lane per (b, d, n); 16-lane shfl_xor reduce for h.C.
// Output fused with Dskip skip-connection and silu(z) gate.
// ---------------------------------------------------------------------------
__global__ void scan_kernel(const float* __restrict__ dt,
                            const float* __restrict__ dbl,
                            const float* __restrict__ xc,
                            const float* __restrict__ xz,
                            const float* __restrict__ A_log,
                            const float* __restrict__ Dskip,
                            float* __restrict__ yg) {
  int g = blockIdx.x * blockDim.x + threadIdx.x;
  if (g >= Bc * DIc * 16) return;
  int n  = g & 15;
  int ch = g >> 4;
  int d  = ch % DIc;
  int b  = ch / DIc;
  float An = -expf(A_log[d * 16 + n]);
  float Dv = Dskip[d];
  float h = 0.f;
  for (int s = 0; s < Sc; ++s) {
    size_t bs = (size_t)b * Sc + s;
    float dtv = dt[bs * DIc + d];
    float x   = xc[bs * DIc + d];
    float Bn  = dbl[bs * 1568 + 1536 + n];
    float Cn  = dbl[bs * 1568 + 1552 + n];
    h = expf(dtv * An) * h + dtv * Bn * x;
    float p = h * Cn;
    p += __shfl_xor(p, 1, 32);
    p += __shfl_xor(p, 2, 32);
    p += __shfl_xor(p, 4, 32);
    p += __shfl_xor(p, 8, 32);
    if (n == 0) {
      float z = xz[bs * 3072 + 1536 + d];
      yg[bs * DIc + d] = (p + Dv * x) * siluf(z);
    }
  }
}

// ---------------------------------------------------------------------------
// RBF phi: one block per token, reduce squared distances to 5 inducing pts.
// ---------------------------------------------------------------------------
__global__ __launch_bounds__(256) void phi_kernel(const float* __restrict__ x,
                                                  const float* __restrict__ ind,
                                                  float* __restrict__ phi) {
  __shared__ float red[256];
  __shared__ float res[NINDc];
  int bs = blockIdx.x;
  int t = threadIdx.x;
  float part[NINDc];
#pragma unroll
  for (int j = 0; j < NINDc; ++j) part[j] = 0.f;
  for (int d = t; d < Dc; d += 256) {
    float xv = x[(size_t)bs * Dc + d];
#pragma unroll
    for (int j = 0; j < NINDc; ++j) {
      float diff = xv - ind[j * Dc + d];
      part[j] += diff * diff;
    }
  }
  for (int j = 0; j < NINDc; ++j) {
    red[t] = part[j];
    __syncthreads();
    for (int o = 128; o > 0; o >>= 1) { if (t < o) red[t] += red[t + o]; __syncthreads(); }
    if (t == 0) res[j] = expf(-0.5f * red[0] / (float)Dc);
    __syncthreads();
  }
  if (t < NINDc) phi[bs * 8 + t] = res[t];
}

__global__ void mean_kernel(const float* __restrict__ x, const float* __restrict__ phi,
                            const float* __restrict__ Wmean, float* __restrict__ out) {
  int idx = blockIdx.x * blockDim.x + threadIdx.x;
  if (idx >= BSc * Dc) return;
  int d = idx % Dc, bs = idx / Dc;
  float acc = x[idx];
#pragma unroll
  for (int j = 0; j < NINDc; ++j) acc += phi[bs * 8 + j] * Wmean[j * Dc + d];
  out[idx] = acc;
}

__global__ void var_kernel(const float* __restrict__ phi, const float* __restrict__ Wvar,
                           float* __restrict__ out) {
  size_t idx = (size_t)blockIdx.x * blockDim.x + threadIdx.x;
  if (idx >= (size_t)BSc * VOCABc) return;
  int v = (int)(idx % VOCABc);
  size_t bs = idx / VOCABc;
  float acc = 0.f;
#pragma unroll
  for (int j = 0; j < NINDc; ++j) acc += phi[bs * 8 + j] * Wvar[(size_t)j * VOCABc + v];
  out[idx] = softplusf(acc);
}

// ---------------------------------------------------------------------------
// Residual + LayerNorm: one block per token. bsum may be null (no residual).
// Safe in-place (each element read/written by exactly one thread).
// ---------------------------------------------------------------------------
__global__ __launch_bounds__(256) void ln_kernel(const float* __restrict__ a,
                                                 const float* __restrict__ bsum,
                                                 const float* __restrict__ g,
                                                 const float* __restrict__ be,
                                                 float* __restrict__ out) {
  __shared__ float red[256];
  __shared__ float smean, svar;
  int bs = blockIdx.x;
  int t = threadIdx.x;
  float v[3];
#pragma unroll
  for (int e = 0; e < 3; ++e) {
    int d = t + e * 256;
    v[e] = a[(size_t)bs * Dc + d];
    if (bsum) v[e] += bsum[(size_t)bs * Dc + d];
  }
  red[t] = v[0] + v[1] + v[2];
  __syncthreads();
  for (int o = 128; o > 0; o >>= 1) { if (t < o) red[t] += red[t + o]; __syncthreads(); }
  if (t == 0) smean = red[0] / (float)Dc;
  __syncthreads();
  float m = smean;
  float sq = 0.f;
#pragma unroll
  for (int e = 0; e < 3; ++e) { float dd = v[e] - m; sq += dd * dd; }
  red[t] = sq;
  __syncthreads();
  for (int o = 128; o > 0; o >>= 1) { if (t < o) red[t] += red[t + o]; __syncthreads(); }
  if (t == 0) svar = red[0] / (float)Dc;
  __syncthreads();
  float rs = rsqrtf(svar + EPSc);
#pragma unroll
  for (int e = 0; e < 3; ++e) {
    int d = t + e * 256;
    out[(size_t)bs * Dc + d] = g[d] * (v[e] - m) * rs + be[d];
  }
}

// ---------------------------------------------------------------------------
// Attention scores + softmax: one block per (b,h,q) row.
// ---------------------------------------------------------------------------
__global__ __launch_bounds__(256) void attn_scores_kernel(const float* __restrict__ Q,
                                                          const float* __restrict__ Kb,
                                                          const int* __restrict__ mask,
                                                          float* __restrict__ P) {
  __shared__ float qv[64];
  __shared__ float sc[512];
  __shared__ float red[256];
  __shared__ float sm, ssum;
  int blk = blockIdx.x;              // ((b*H + h)*S + q)
  int q = blk % Sc;
  int h = (blk / Sc) % Hc;
  int b = blk / (Sc * Hc);
  int t = threadIdx.x;
  if (t < 64) qv[t] = Q[((size_t)(b * Sc + q)) * Dc + h * HDc + t];
  __syncthreads();
#pragma unroll
  for (int r = 0; r < 2; ++r) {
    int k = t + r * 256;
    const float* kp = &Kb[((size_t)(b * Sc + k)) * Dc + h * HDc];
    float dot = 0.f;
    for (int dd = 0; dd < 64; ++dd) dot += qv[dd] * kp[dd];
    float bv = (mask[b * Sc + k] == 0) ? -1e9f : 0.f;
    sc[k] = dot * 0.125f + bv;       // 1/sqrt(64)
  }
  __syncthreads();
  red[t] = fmaxf(sc[t], sc[t + 256]);
  __syncthreads();
  for (int o = 128; o > 0; o >>= 1) { if (t < o) red[t] = fmaxf(red[t], red[t + o]); __syncthreads(); }
  if (t == 0) sm = red[0];
  __syncthreads();
  float e0 = expf(sc[t] - sm), e1 = expf(sc[t + 256] - sm);
  sc[t] = e0; sc[t + 256] = e1;
  red[t] = e0 + e1;
  __syncthreads();
  for (int o = 128; o > 0; o >>= 1) { if (t < o) red[t] += red[t + o]; __syncthreads(); }
  if (t == 0) ssum = red[0];
  __syncthreads();
  float inv = 1.f / ssum;
  float* pp = &P[(size_t)blk * Sc];
  pp[t] = sc[t] * inv;
  pp[t + 256] = sc[t + 256] * inv;
}

__global__ void attn_ctx_kernel(const float* __restrict__ P, const float* __restrict__ V,
                                float* __restrict__ ctx) {
  int idx = blockIdx.x * blockDim.x + threadIdx.x;
  if (idx >= BSc * Dc) return;
  int dh = idx & 63;
  int h = (idx >> 6) % Hc;
  int bs = idx / Dc;
  int b = bs / Sc, q = bs % Sc;
  const float* pp = &P[(((size_t)(b * Hc + h)) * Sc + q) * Sc];
  float acc = 0.f;
  for (int k = 0; k < Sc; ++k)
    acc += pp[k] * V[((size_t)(b * Sc + k)) * Dc + h * HDc + dh];
  ctx[idx] = acc;
}

// ---------------------------------------------------------------------------
// KAN activation: silu(h) + sum_g exp(-(h - grid_g)^2) * coeff[f][g]
// ---------------------------------------------------------------------------
__global__ void kan_act_kernel(const float* __restrict__ hb, const float* __restrict__ coeff,
                               float* __restrict__ act) {
  int idx = blockIdx.x * blockDim.x + threadIdx.x;
  if (idx >= BSc * DFFc) return;
  int f = idx % DFFc;
  float hv = hb[idx];
  float s = siluf(hv);
#pragma unroll
  for (int g8 = 0; g8 < 8; ++g8) {
    float gg = -2.f + 4.f * (float)g8 / 7.f;
    float diff = hv - gg;
    s += expf(-diff * diff) * coeff[f * 8 + g8];
  }
  act[idx] = s;
}

// ---------------------------------------------------------------------------
// Host launcher
// ---------------------------------------------------------------------------
extern "C" void kernel_launch(void* const* d_in, const int* in_sizes, int n_in,
                              void* d_out, int out_size, void* d_ws, size_t ws_size,
                              hipStream_t stream) {
  (void)in_sizes; (void)n_in; (void)out_size; (void)ws_size;
  auto PF = [&](int i) { return (const float*)d_in[i]; };
  const float* inp  = PF(0);
  const int*   mask = (const int*)d_in[1];
  // flattening: 0 inputs, 1 mask, 2..33 layer0, 34..65 layer1,
  // 66 inducing, 67 Wmean, 68 Wvar, 69 final_g, 70 final_b
  const float* inducing = PF(66);
  const float* Wmean    = PF(67);
  const float* Wvar     = PF(68);
  const float* fg       = PF(69);
  const float* fb       = PF(70);

  float* ws = (float*)d_ws;
  size_t off = 0;
  auto alloc = [&](size_t n) { float* p = ws + off; off += (n + 255) & ~(size_t)255; return p; };
  float* X   = alloc((size_t)BSc * Dc);        // residual stream
  float* xz  = alloc((size_t)BSc * 3072);
  float* xc  = alloc((size_t)BSc * DIc);
  float* dbl = alloc((size_t)BSc * 1568);
  float* dtb = alloc((size_t)BSc * DIc);
  float* yg  = alloc((size_t)BSc * DIc);
  float* x2  = alloc((size_t)BSc * Dc);
  float* mn  = alloc((size_t)BSc * Dc);
  float* phi = alloc((size_t)BSc * 8);
  float* qb  = alloc((size_t)BSc * Dc);
  float* kb  = alloc((size_t)BSc * Dc);
  float* vb  = alloc((size_t)BSc * Dc);
  float* Pb  = alloc((size_t)Bc * Hc * Sc * Sc);
  float* cx  = alloc((size_t)BSc * Dc);
  float* t1  = alloc((size_t)BSc * Dc);
  float* fh  = alloc((size_t)BSc * DFFc);
  float* ab  = alloc((size_t)BSc * DFFc);

  auto gemm = [&](const float* A, int lda, const float* W, int ldw, const float* bias,
                  float* C, int ldc, int M, int N, int K, int epi) {
    dim3 g((N + 63) / 64, (M + 127) / 128);
    gemm_bf16_wmma<<<g, dim3(256), 0, stream>>>(A, lda, W, ldw, bias, C, ldc, M, N, K, epi);
  };

  rotary_kernel<<<(BSc * Dc + 255) / 256, 256, 0, stream>>>(inp, X);

  for (int l = 0; l < 2; ++l) {
    int base = 2 + l * 32;
    const float *Win = PF(base + 0), *cw = PF(base + 1), *cbv = PF(base + 2);
    const float *Wx = PF(base + 3), *Wdt = PF(base + 4), *bdt = PF(base + 5);
    const float *Alog = PF(base + 6), *Dsk = PF(base + 7), *Wout = PF(base + 8);
    const float *pW = PF(base + 9), *pb = PF(base + 10);
    const float *Wq = PF(base + 11), *bq = PF(base + 12), *Wk = PF(base + 13), *bkv = PF(base + 14);
    const float *Wv = PF(base + 15), *bvv = PF(base + 16), *Wo = PF(base + 17), *bo = PF(base + 18);
    const float *g1 = PF(base + 19), *be1 = PF(base + 20), *g2 = PF(base + 21), *be2 = PF(base + 22);
    const float *W1 = PF(base + 23), *b1 = PF(base + 24), *W2 = PF(base + 25), *b2 = PF(base + 26);
    const float *kW1 = PF(base + 27), *kb1 = PF(base + 28), *kco = PF(base + 29);
    const float *kW2 = PF(base + 30), *kb2 = PF(base + 31);

    // ---- Mamba ----
    gemm(X, Dc, Win, 3072, nullptr, xz, 3072, BSc, 3072, Dc, EPI_NONE);
    conv_silu_kernel<<<(BSc * DIc + 255) / 256, 256, 0, stream>>>(xz, cw, cbv, xc);
    gemm(xc, DIc, Wx, 1568, nullptr, dbl, 1568, BSc, 1568, DIc, EPI_NONE);
    gemm(dbl, 1568, Wdt, DIc, bdt, dtb, DIc, BSc, DIc, DIc, EPI_SOFTPLUS);
    scan_kernel<<<(Bc * DIc * 16 + 255) / 256, 256, 0, stream>>>(dtb, dbl, xc, xz, Alog, Dsk, yg);
    gemm(yg, DIc, Wout, Dc, nullptr, x2, Dc, BSc, Dc, DIc, EPI_NONE);

    // ---- Uncertainty (mean path; phi kept for final var) ----
    phi_kernel<<<BSc, 256, 0, stream>>>(x2, inducing, phi);
    mean_kernel<<<(BSc * Dc + 255) / 256, 256, 0, stream>>>(x2, phi, Wmean, mn);

    // ---- Projection ----
    gemm(mn, Dc, pW, Dc, pb, X, Dc, BSc, Dc, Dc, EPI_NONE);

    // ---- Attention block ----
    gemm(X, Dc, Wq, Dc, bq, qb, Dc, BSc, Dc, Dc, EPI_NONE);
    gemm(X, Dc, Wk, Dc, bkv, kb, Dc, BSc, Dc, Dc, EPI_NONE);
    gemm(X, Dc, Wv, Dc, bvv, vb, Dc, BSc, Dc, Dc, EPI_NONE);
    attn_scores_kernel<<<Bc * Hc * Sc, 256, 0, stream>>>(qb, kb, mask, Pb);
    attn_ctx_kernel<<<(BSc * Dc + 255) / 256, 256, 0, stream>>>(Pb, vb, cx);
    gemm(cx, Dc, Wo, Dc, bo, t1, Dc, BSc, Dc, Dc, EPI_NONE);
    ln_kernel<<<BSc, 256, 0, stream>>>(X, t1, g1, be1, X);
    gemm(X, Dc, W1, DFFc, b1, fh, DFFc, BSc, DFFc, Dc, EPI_GELU);
    gemm(fh, DFFc, W2, Dc, b2, t1, Dc, BSc, Dc, DFFc, EPI_NONE);
    ln_kernel<<<BSc, 256, 0, stream>>>(X, t1, g2, be2, X);

    // ---- KAN ----
    gemm(X, Dc, kW1, DFFc, kb1, fh, DFFc, BSc, DFFc, Dc, EPI_NONE);
    kan_act_kernel<<<(BSc * DFFc + 255) / 256, 256, 0, stream>>>(fh, kco, ab);
    gemm(ab, DFFc, kW2, Dc, kb2, X, Dc, BSc, Dc, DFFc, EPI_NONE);
  }

  // final layernorm -> first output section
  ln_kernel<<<BSc, 256, 0, stream>>>(X, nullptr, fg, fb, (float*)d_out);
  // variance from last layer's phi -> second output section
  size_t nvar = (size_t)BSc * VOCABc;
  var_kernel<<<(unsigned)((nvar + 255) / 256), 256, 0, stream>>>(
      phi, Wvar, (float*)d_out + (size_t)BSc * Dc);
}
